// CusGATP_Net_38397007626855
// MI455X (gfx1250) — matrix operand order repaired and checked by reading, not compiled, not run on hardware
//
#include <hip/hip_runtime.h>
#include <math.h>

// ---------------------------------------------------------------------------
// CusGATP net for MI455X (gfx1250, wave32, WMMA + async tensor data path).
// Dense algebra -> one WMMA GEMM kernel:
//   * 128x64 tile, 8 waves, 4 x v_wmma_f32_16x16x32_f16 per wave per K-step
//   * double-buffered LDS staged by GLOBAL_LOAD_ASYNC_TO_LDS_B128 (ASYNCcnt),
//     issued for tile t+1 while tile t is computed; s_wait_asynccnt + barrier
//   * LDS tiles are f32 (async mover cannot convert); f32->f16 happens at
//     fragment assembly; edge tiles use clamped synchronous loads
// Irregular ops (line pooling, gathers, segment softmax, symmetric scatter)
// are VALU kernels.
//
// Input flattening (JAX pytree: dict keys sorted at every level):
//   0 edge_attr, 1 edge_index(int32), 2 features,
//   3..16  params['conv_in']  (Wa,We,Wn,Wpi,Wpj,ba,be,bn,bpi,bpj,
//                              fc_edge.W,fc_edge.b,fc_pt.W,fc_pt.b)
//  17..30  params['conv_out'], 31,32 fc_edge, 33..38 fc_hour1..3,
//  39,40 fc_loi, 41,42 fc_node, 43..56 hidden0, 57..70 hidden1, 71 x.
// Output: [ out_node (1024x64) | sym_edges (8192x2) ] float32.
// ---------------------------------------------------------------------------

typedef __attribute__((ext_vector_type(16))) _Float16 v16h;
typedef __attribute__((ext_vector_type(8)))  float    v8f;

#define NN      1024
#define NE      8192
#define CF_     256
#define HW_     16384
#define DLOI    128
#define NPTS0   32
#define HID_    128
#define EHID_   128
#define HFC_    512
#define HOUT_   128
#define HEADS_  4
#define ODIM_   64
#define EOUT_   2

enum {
  IN_EDGE_ATTR = 0, IN_EDGE_INDEX = 1, IN_FEATURES = 2,
  P_CONV_IN = 3, P_CONV_OUT = 17,
  P_FCEDGE_W = 31, P_FCEDGE_B = 32,
  P_H1_W = 33, P_H1_B = 34, P_H2_W = 35, P_H2_B = 36, P_H3_W = 37, P_H3_B = 38,
  P_FCLOI_W = 39, P_FCLOI_B = 40, P_FCNODE_W = 41, P_FCNODE_B = 42,
  P_HIDDEN0 = 43, P_HIDDEN1 = 57, IN_X = 71
};
enum { G_Wa=0, G_We=1, G_Wn=2, G_Wpi=3, G_Wpj=4, G_ba=5, G_be=6, G_bn=7,
       G_bpi=8, G_bpj=9, G_FCE_W=10, G_FCE_B=11, G_FCP_W=12, G_FCP_B=13 };

// ============================ WMMA GEMM =====================================
// C[m,n] = act( sum_k A[m*As_m + k] * B[k*Bs_k + n] + bias[n] ),  C[m*Cs_m+n].

#define GT_M 128
#define GT_N 64
#define GT_K 32
#define LDA  36      // A row pitch (floats): 144B, 16B-aligned
#define LDB  68      // B row pitch (floats): 272B, 16B-aligned
#define GEMM_THREADS 256

__device__ __forceinline__ float gemm_act(float v, int act) {
  if (act == 1) return v > 0.f ? v : 0.f;          // relu
  if (act == 2) return v > 0.f ? v : 0.2f * v;     // leaky_relu(0.2)
  if (act == 3) return 1.f / (1.f + expf(-v));     // sigmoid
  return v;
}

// Async copy of 16 bytes global -> LDS (GLOBAL_LOAD_ASYNC_TO_LDS_B128,
// tracked by ASYNCcnt). lds_off = LDS byte address (low 32 bits of the
// shared-memory pointer, per ISA 10.2 aperture truncation).
__device__ __forceinline__ void async_copy_b128(void* lds_ptr, const float* gsrc) {
  unsigned lds_off = (unsigned)(unsigned long long)lds_ptr;
  asm volatile("global_load_async_to_lds_b128 %0, %1, off"
               :: "v"(lds_off), "v"(gsrc) : "memory");
}
__device__ __forceinline__ void wait_asynccnt0() {
  asm volatile("s_wait_asynccnt 0x0" ::: "memory");
}

__global__ __launch_bounds__(GEMM_THREADS)
void wmma_gemm_kernel(const float* __restrict__ A, long long As_m,
                      const float* __restrict__ B, long long Bs_k,
                      const float* __restrict__ bias,
                      float* __restrict__ C, long long Cs_m,
                      int M, int N, int K, int act)
{
  __shared__ float sA[2][GT_M][LDA];   // [buf][m][k]   (f32)
  __shared__ float sB[2][GT_K][LDB];   // [buf][k][n]   (f32)

  const int tid  = threadIdx.x;
  const int lane = tid & 31;
  const int wave = tid >> 5;       // 0..7
  const int wm   = wave >> 1;      // 0..3 -> 32-row quadrant
  const int wn   = wave & 1;       // 0..1 -> 32-col quadrant
  const int half = lane >> 4;
  const int r    = lane & 15;

  const long long m0 = (long long)blockIdx.y * GT_M;
  const long long n0 = (long long)blockIdx.x * GT_N;

  // Block-uniform fast-path eligibility (scalar branch, no divergence).
  const bool blk_fast = (m0 + GT_M <= M) && (n0 + GT_N <= N) &&
                        ((As_m & 3LL) == 0LL) && ((Bs_k & 3LL) == 0LL);
  const int ntiles = (K + GT_K - 1) / GT_K;

  // Stage tile t into buffer buf. Full interior tiles use the async data
  // mover (no VGPR round-trip); ragged tiles use clamped synchronous loads.
  auto stage = [&](int t, int buf) {
    const int k0 = t * GT_K;
    if (blk_fast && (k0 + GT_K <= K)) {
      for (int i = tid; i < GT_M * (GT_K / 4); i += GEMM_THREADS) {   // 4/thr
        const int mm = i >> 3, kg = (i & 7) * 4;
        async_copy_b128(&sA[buf][mm][kg], A + (m0 + mm) * As_m + (k0 + kg));
      }
      for (int i = tid; i < GT_K * (GT_N / 4); i += GEMM_THREADS) {   // 2/thr
        const int kk = i >> 4, ng = (i & 15) * 4;
        async_copy_b128(&sB[buf][kk][ng], B + (long long)(k0 + kk) * Bs_k + (n0 + ng));
      }
    } else {
      for (int i = tid; i < GT_M * GT_K; i += GEMM_THREADS) {
        const int mm = i >> 5, kk = i & 31;
        const long long gm = m0 + mm;
        const long long gmc = gm < M ? gm : (long long)M - 1;
        const int gk = k0 + kk;
        const int gkc = gk < K ? gk : K - 1;
        float a = A[gmc * As_m + gkc];
        if (!(gm < M && gk < K)) a = 0.f;
        sA[buf][mm][kk] = a;
      }
      for (int i = tid; i < GT_K * GT_N; i += GEMM_THREADS) {
        const int kk = i >> 6, nn = i & 63;
        const long long gn = n0 + nn;
        const long long gnc = gn < N ? gn : (long long)N - 1;
        const int gk = k0 + kk;
        const int gkc = gk < K ? gk : K - 1;
        float b = B[(long long)gkc * Bs_k + gnc];
        if (!(gn < N && gk < K)) b = 0.f;
        sB[buf][kk][nn] = b;
      }
    }
  };

  v8f acc00 = {}, acc01 = {}, acc10 = {}, acc11 = {};

  stage(0, 0);                       // prologue: tile 0 in flight
  for (int t = 0; t < ntiles; ++t) {
    const int cur = t & 1;
    wait_asynccnt0();                // this wave's async stores to LDS done
    __syncthreads();                 // whole tile visible to all waves
    if (t + 1 < ntiles) stage(t + 1, 1 - cur);   // overlap next tile w/ math

    // ---- fragments per ISA 7.12.2 (16-bit A 16x32 / B 32x16 layouts) ----
    v16h af0, af1, bf0, bf1;
#pragma unroll
    for (int v = 0; v < 8; ++v) {
      const int ka = (v < 4 ? 2 * v : 16 + 2 * (v - 4)) + half * 8;
      const float2 a0 = *(const float2*)&sA[cur][wm * 32 + r][ka];
      const float2 a1 = *(const float2*)&sA[cur][wm * 32 + 16 + r][ka];
      af0[2 * v] = (_Float16)a0.x; af0[2 * v + 1] = (_Float16)a0.y;
      af1[2 * v] = (_Float16)a1.x; af1[2 * v + 1] = (_Float16)a1.y;
      const int kb = (v < 4 ? 2 * v : 16 + 2 * (v - 4)) + half * 8;
      bf0[2 * v]     = (_Float16)sB[cur][kb][wn * 32 + r];
      bf0[2 * v + 1] = (_Float16)sB[cur][kb + 1][wn * 32 + r];
      bf1[2 * v]     = (_Float16)sB[cur][kb][wn * 32 + 16 + r];
      bf1[2 * v + 1] = (_Float16)sB[cur][kb + 1][wn * 32 + 16 + r];
    }
    acc00 = __builtin_amdgcn_wmma_f32_16x16x32_f16(false, af0, false, bf0, (short)0, acc00, false, false);
    acc01 = __builtin_amdgcn_wmma_f32_16x16x32_f16(false, af0, false, bf1, (short)0, acc01, false, false);
    acc10 = __builtin_amdgcn_wmma_f32_16x16x32_f16(false, af1, false, bf0, (short)0, acc10, false, false);
    acc11 = __builtin_amdgcn_wmma_f32_16x16x32_f16(false, af1, false, bf1, (short)0, acc11, false, false);
    // No trailing barrier needed: writes into buffer `cur` happen two tiles
    // later, after the top-of-iteration barrier, and each wave's fragment
    // reads complete before it can reach that barrier (WMMA waits DScnt).
  }

  // ---- epilogue: C/D layout (VGPR i -> M = i + 8*half, lane -> N) ----
#pragma unroll
  for (int i = 0; i < 8; ++i) {
    const long long gmA = m0 + wm * 32 + half * 8 + i;
    const long long gmB = gmA + 16;
    const long long gn0 = n0 + wn * 32 + r;
    const long long gn1 = gn0 + 16;
    float vb0 = (gn0 < N && bias) ? bias[gn0] : 0.f;
    float vb1 = (gn1 < N && bias) ? bias[gn1] : 0.f;
    if (gmA < M) {
      if (gn0 < N) C[gmA * Cs_m + gn0] = gemm_act(acc00[i] + vb0, act);
      if (gn1 < N) C[gmA * Cs_m + gn1] = gemm_act(acc01[i] + vb1, act);
    }
    if (gmB < M) {
      if (gn0 < N) C[gmB * Cs_m + gn0] = gemm_act(acc10[i] + vb0, act);
      if (gn1 < N) C[gmB * Cs_m + gn1] = gemm_act(acc11[i] + vb1, act);
    }
  }
}

static inline void gemm(hipStream_t s,
                        const float* A, long long As_m,
                        const float* B, long long Bs_k,
                        const float* bias, float* C, long long Cs_m,
                        int M, int N, int K, int act)
{
  dim3 g((N + GT_N - 1) / GT_N, (M + GT_M - 1) / GT_M);
  wmma_gemm_kernel<<<g, GEMM_THREADS, 0, s>>>(A, As_m, B, Bs_k, bias, C, Cs_m,
                                              M, N, K, act);
}

// ======================= irregular / VALU kernels ==========================

__global__ void fill_kernel(float* p, float v, long long n) {
  for (long long i = (long long)blockIdx.x * blockDim.x + threadIdx.x; i < n;
       i += (long long)gridDim.x * blockDim.x) p[i] = v;
}
__global__ void fill_int_kernel(int* p, int v, long long n) {
  for (long long i = (long long)blockIdx.x * blockDim.x + threadIdx.x; i < n;
       i += (long long)gridDim.x * blockDim.x) p[i] = v;
}

// features (256 x 16384) -> featT (16384 x 256), LDS-tiled transpose
__global__ __launch_bounds__(256)
void transpose_feat_kernel(const float* __restrict__ f, float* __restrict__ ft) {
  __shared__ float tile[32][33];
  const int hw0 = blockIdx.x * 32, c0 = blockIdx.y * 32;
  const int tx = threadIdx.x & 31, ty = threadIdx.x >> 5;  // ty 0..7
  for (int yy = ty; yy < 32; yy += 8)
    tile[yy][tx] = f[(long long)(c0 + yy) * HW_ + hw0 + tx];
  __syncthreads();
  for (int yy = ty; yy < 32; yy += 8)
    ft[(long long)(hw0 + yy) * CF_ + c0 + tx] = tile[tx][yy];
}

__global__ void gather_rows_kernel(const float* __restrict__ src,
                                   const int* __restrict__ idx,
                                   float* __restrict__ dst,
                                   int E, int width, long long s_stride,
                                   long long d_stride, long long d_off) {
  long long n = (long long)E * width;
  for (long long i = (long long)blockIdx.x * blockDim.x + threadIdx.x; i < n;
       i += (long long)gridDim.x * blockDim.x) {
    long long e = i / width, j = i % width;
    dst[e * d_stride + d_off + j] = src[(long long)idx[e] * s_stride + j];
  }
}

__global__ void copy2d_kernel(const float* __restrict__ src, long long s_stride,
                              float* __restrict__ dst, long long d_stride,
                              long long d_off, int rows, int width) {
  long long n = (long long)rows * width;
  for (long long i = (long long)blockIdx.x * blockDim.x + threadIdx.x; i < n;
       i += (long long)gridDim.x * blockDim.x) {
    long long r = i / width, j = i % width;
    dst[r * d_stride + d_off + j] = src[r * s_stride + j];
  }
}

__global__ void bcast_edge_kernel(const float* __restrict__ ea, int ein,
                                  float* __restrict__ dst, int W, int off,
                                  int E, int H) {
  long long n = (long long)E * H * ein;
  for (long long i = (long long)blockIdx.x * blockDim.x + threadIdx.x; i < n;
       i += (long long)gridDim.x * blockDim.x) {
    long long e = i / ((long long)H * ein);
    long long rem = i % ((long long)H * ein);
    long long h = rem / ein, j = rem % ein;
    dst[(e * H + h) * W + off + j] = ea[e * ein + j];
  }
}

__global__ void gather_head_kernel(const float* __restrict__ upd,
                                   const int* __restrict__ idx,
                                   float* __restrict__ dst,
                                   int E, int H, int c, int W, int off) {
  long long n = (long long)E * H * c;
  for (long long i = (long long)blockIdx.x * blockDim.x + threadIdx.x; i < n;
       i += (long long)gridDim.x * blockDim.x) {
    long long e = i / ((long long)H * c);
    long long rem = i % ((long long)H * c);
    long long h = rem / c, j = rem % c;
    dst[(e * H + h) * W + off + j] = upd[((long long)idx[e] * H + h) * c + j];
  }
}

__device__ __forceinline__ void atomicMaxF(float* addr, float v) {
  unsigned* ua = (unsigned*)addr;
  unsigned old = *ua;
  while (true) {
    float f = __uint_as_float(old);
    if (f >= v) break;
    unsigned prev = atomicCAS(ua, old, __float_as_uint(v));
    if (prev == old) break;
    old = prev;
  }
}

__global__ void seg_max_kernel(const float* __restrict__ alpha,
                               const int* __restrict__ col,
                               float* __restrict__ amax, int E, int HC) {
  long long n = (long long)E * HC;
  for (long long i = (long long)blockIdx.x * blockDim.x + threadIdx.x; i < n;
       i += (long long)gridDim.x * blockDim.x) {
    long long e = i / HC, t = i % HC;
    atomicMaxF(&amax[(long long)col[e] * HC + t], alpha[i]);
  }
}

__global__ void fix_nonfinite_kernel(float* p, long long n) {
  for (long long i = (long long)blockIdx.x * blockDim.x + threadIdx.x; i < n;
       i += (long long)gridDim.x * blockDim.x) {
    float v = p[i];
    if (!(fabsf(v) <= 3.402823466e38f)) p[i] = 0.f;
  }
}

__global__ void exp_denom_kernel(float* __restrict__ alpha,
                                 const float* __restrict__ amax,
                                 const int* __restrict__ col,
                                 float* __restrict__ denom, int E, int HC) {
  long long n = (long long)E * HC;
  for (long long i = (long long)blockIdx.x * blockDim.x + threadIdx.x; i < n;
       i += (long long)gridDim.x * blockDim.x) {
    long long e = i / HC, t = i % HC;
    long long g = (long long)col[e] * HC + t;
    float v = expf(alpha[i] - amax[g]);
    alpha[i] = v;
    atomicAdd(&denom[g], v);
  }
}

__global__ void upd_kernel(const float* __restrict__ ex,
                           const float* __restrict__ denom,
                           const float* __restrict__ xj,
                           const int* __restrict__ col,
                           float* __restrict__ upd, int E, int HC) {
  long long n = (long long)E * HC;
  for (long long i = (long long)blockIdx.x * blockDim.x + threadIdx.x; i < n;
       i += (long long)gridDim.x * blockDim.x) {
    long long e = i / HC, t = i % HC;
    long long g = (long long)col[e] * HC + t;
    atomicAdd(&upd[g], ex[i] / (denom[g] + 1e-16f) * xj[i]);
  }
}

__global__ __launch_bounds__(DLOI)
void pooling_sample_kernel(const float* __restrict__ loi,   // [hw][128]
                           const float* __restrict__ x,
                           const int* __restrict__ row,
                           const int* __restrict__ col,
                           float* __restrict__ feat)        // [E][2048]
{
  const int e = blockIdx.x;
  const int c = threadIdx.x;
  const float w = 128.f;
  const int ri = row[e], ci = col[e];
  const float Ux = x[ri * 2] * w, Uy = x[ri * 2 + 1] * w;
  const float Vx = x[ci * 2] * w, Vy = x[ci * 2 + 1] * w;
  const float dx = Vx - Ux, dy = Vy - Uy;
  const float nrm = fmaxf(sqrtf(dx * dx + dy * dy), 1e-12f);
  const float lnx = dy / nrm, lny = -dx / nrm;   // line_dir @ rot90

  float vals[NPTS0];
#pragma unroll 4
  for (int i = 0; i < NPTS0; ++i) {
    const float t0 = (float)i / (float)(NPTS0 - 1);
    const float dpx = Ux * t0 + Vx * (1.f - t0);
    const float dpy = Uy * t0 + Vy * (1.f - t0);
    float m = -3.4e38f;
    for (int j = 0; j < 3; ++j) {
      const float t1 = (float)(j - 1);
      const float px = dpx + lnx * t1, py = dpy + lny * t1;
      const float px0 = fminf(fmaxf(floorf(px), 0.f), 127.f);
      const float py0 = fminf(fmaxf(floorf(py), 0.f), 127.f);
      const float px1 = fminf(px0 + 1.f, 127.f);
      const float py1 = fminf(py0 + 1.f, 127.f);
      const int ix0 = (int)px0, iy0 = (int)py0, ix1 = (int)px1, iy1 = (int)py1;
      const float s =
          loi[((long long)iy0 * 128 + ix0) * DLOI + c] * (py1 - py) * (px1 - px) +
          loi[((long long)iy1 * 128 + ix0) * DLOI + c] * (py - py0) * (px1 - px) +
          loi[((long long)iy0 * 128 + ix1) * DLOI + c] * (py1 - py) * (px - px0) +
          loi[((long long)iy1 * 128 + ix1) * DLOI + c] * (py - py0) * (px - px0);
      m = fmaxf(m, s);
    }
    vals[i] = m;
  }
#pragma unroll
  for (int t = 0; t < NPTS0 / 2; ++t)
    feat[(long long)e * (DLOI * NPTS0 / 2) + c * (NPTS0 / 2) + t] =
        fmaxf(vals[2 * t], vals[2 * t + 1]);
}

__global__ void winner_kernel(const int* row, const int* col, int* winner, int E) {
  for (int e = blockIdx.x * blockDim.x + threadIdx.x; e < E;
       e += gridDim.x * blockDim.x)
    atomicMax(&winner[(long long)row[e] * NN + col[e]], e);
}
__global__ void mat_init_kernel(float* mat) {
  long long n = (long long)NN * NN;
  for (long long i = (long long)blockIdx.x * blockDim.x + threadIdx.x; i < n;
       i += (long long)gridDim.x * blockDim.x) {
    mat[i * 2] = 1.0f; mat[i * 2 + 1] = 0.0f;
  }
}
__global__ void mat_set_kernel(const float* oe, const int* row, const int* col,
                               const int* winner, float* mat, int E) {
  for (int e = blockIdx.x * blockDim.x + threadIdx.x; e < E;
       e += gridDim.x * blockDim.x) {
    long long g = (long long)row[e] * NN + col[e];
    if (winner[g] == e) {
      mat[g * 2]     = oe[(long long)e * 2];
      mat[g * 2 + 1] = oe[(long long)e * 2 + 1];
    }
  }
}
__global__ void sym_kernel(const float* mat, const int* row, const int* col,
                           float* out, int E) {
  for (int e = blockIdx.x * blockDim.x + threadIdx.x; e < E;
       e += gridDim.x * blockDim.x) {
    long long a = (long long)row[e] * NN + col[e];
    long long b = (long long)col[e] * NN + row[e];
    out[(long long)e * 2]     = 0.5f * (mat[a * 2]     + mat[b * 2]);
    out[(long long)e * 2 + 1] = 0.5f * (mat[a * 2 + 1] + mat[b * 2 + 1]);
  }
}

// ============================ host-side glue ===============================

static inline unsigned ew_blocks(long long n) {
  long long b = (n + 255) / 256;
  return (unsigned)(b < 1 ? 1 : (b > 262144 ? 262144 : b));
}
#define EW(n) ew_blocks(n), 256, 0, s

struct GatScratch {
  float *xfc, *xi, *xj, *cat, *alpha, *amax, *denom, *upd, *ue, *pt_cat, *edge_cat;
};

static void run_gat(hipStream_t s, void* const* d_in, int base,
                    int cin, int cout, int ein, int eout,
                    const float* hn, const float* he,
                    float* out_pt, float* out_edge, int edge_act,
                    const int* row, const int* col, const GatScratch& t)
{
  const int H = HEADS_, E = NE, n = NN;
  const int Wc = 2 * cout + ein;
  const int HC = H * cout;
  auto P = [&](int i) { return (const float*)d_in[i]; };
  const float* Wa  = P(base + G_Wa);  const float* ba  = P(base + G_ba);
  const float* We  = P(base + G_We);  const float* be  = P(base + G_be);
  const float* Wn  = P(base + G_Wn);  const float* bn  = P(base + G_bn);
  const float* Wpi = P(base + G_Wpi); const float* bpi = P(base + G_bpi);
  const float* Wpj = P(base + G_Wpj); const float* bpj = P(base + G_bpj);

  // 1) xfc[n,h,o] = hn @ Wn[h] + bn[h]
  for (int h = 0; h < H; ++h)
    gemm(s, hn, cin, Wn + (long long)h * cin * cout, cout,
         bn + (long long)h * cout, t.xfc + (long long)h * cout,
         (long long)H * cout, n, cout, cin, 0);

  // 2) xi = xfc[col], xj = xfc[row]
  gather_rows_kernel<<<EW((long long)E * HC)>>>(t.xfc, col, t.xi, E, HC, HC, HC, 0);
  gather_rows_kernel<<<EW((long long)E * HC)>>>(t.xfc, row, t.xj, E, HC, HC, HC, 0);

  // 3) phi_i / phi_j straight into cat slices; 4) ea broadcast
  for (int h = 0; h < H; ++h) {
    gemm(s, t.xi + (long long)h * cout, HC,
         Wpi + (long long)h * cout * cout, cout, bpi + (long long)h * cout,
         t.cat + (long long)h * Wc, (long long)H * Wc, E, cout, cout, 0);
    gemm(s, t.xj + (long long)h * cout, HC,
         Wpj + (long long)h * cout * cout, cout, bpj + (long long)h * cout,
         t.cat + (long long)h * Wc + cout, (long long)H * Wc, E, cout, cout, 0);
  }
  bcast_edge_kernel<<<EW((long long)E * H * ein)>>>(he, ein, t.cat, Wc, 2 * cout, E, H);

  // 5) alpha = leaky_relu(cat @ Wa + ba)
  for (int h = 0; h < H; ++h)
    gemm(s, t.cat + (long long)h * Wc, (long long)H * Wc,
         Wa + (long long)h * Wc * cout, cout, ba + (long long)h * cout,
         t.alpha + (long long)h * cout, (long long)HC, E, cout, Wc, 2);

  // 6) segment softmax over col + weighted aggregation of xj
  fill_kernel<<<EW((long long)n * HC)>>>(t.amax, -3.4e38f, (long long)n * HC);
  seg_max_kernel<<<EW((long long)E * HC)>>>(t.alpha, col, t.amax, E, HC);
  fix_nonfinite_kernel<<<EW((long long)n * HC)>>>(t.amax, (long long)n * HC);
  fill_kernel<<<EW((long long)n * HC)>>>(t.denom, 0.f, (long long)n * HC);
  exp_denom_kernel<<<EW((long long)E * HC)>>>(t.alpha, t.amax, col, t.denom, E, HC);
  fill_kernel<<<EW((long long)n * HC)>>>(t.upd, 0.f, (long long)n * HC);
  upd_kernel<<<EW((long long)E * HC)>>>(t.alpha, t.denom, t.xj, col, t.upd, E, HC);

  // 7) ecat = [upd[row], upd[col], ea_h]  (reuses cat buffer)
  gather_head_kernel<<<EW((long long)E * HC)>>>(t.upd, row, t.cat, E, H, cout, Wc, 0);
  gather_head_kernel<<<EW((long long)E * HC)>>>(t.upd, col, t.cat, E, H, cout, Wc, cout);
  bcast_edge_kernel<<<EW((long long)E * H * ein)>>>(he, ein, t.cat, Wc, 2 * cout, E, H);

  // 8) ue = ecat @ We + be
  for (int h = 0; h < H; ++h)
    gemm(s, t.cat + (long long)h * Wc, (long long)H * Wc,
         We + (long long)h * Wc * eout, eout, be + (long long)h * eout,
         t.ue + (long long)h * eout, (long long)H * eout, E, eout, Wc, 0);

  // 9) out_pt = [hn, upd] @ fc_pt.W + b
  const int Kp = cin + HC;
  copy2d_kernel<<<EW((long long)n * cin)>>>(hn, cin, t.pt_cat, Kp, 0, n, cin);
  copy2d_kernel<<<EW((long long)n * HC)>>>(t.upd, HC, t.pt_cat, Kp, cin, n, HC);
  gemm(s, t.pt_cat, Kp, P(base + G_FCP_W), cout, P(base + G_FCP_B),
       out_pt, cout, n, cout, Kp, 0);

  // 10) out_edge = [he, ue] @ fc_edge.W + b (sigmoid only on final layer)
  const int Ke = ein + H * eout;
  copy2d_kernel<<<EW((long long)E * ein)>>>(he, ein, t.edge_cat, Ke, 0, E, ein);
  copy2d_kernel<<<EW((long long)E * H * eout)>>>(t.ue, (long long)H * eout,
                                                 t.edge_cat, Ke, ein, E, H * eout);
  gemm(s, t.edge_cat, Ke, P(base + G_FCE_W), eout, P(base + G_FCE_B),
       out_edge, eout, E, eout, Ke, edge_act);
}

extern "C" void kernel_launch(void* const* d_in, const int* in_sizes, int n_in,
                              void* d_out, int out_size, void* d_ws, size_t ws_size,
                              hipStream_t stream)
{
  hipStream_t s = stream;
  const float* edge_attr = (const float*)d_in[IN_EDGE_ATTR];
  const int*   eidx      = (const int*)d_in[IN_EDGE_INDEX];
  const float* features  = (const float*)d_in[IN_FEATURES];
  const float* x         = (const float*)d_in[IN_X];
  const int* row = eidx;
  const int* col = eidx + NE;
  float* out = (float*)d_out;
  auto P = [&](int i) { return (const float*)d_in[i]; };

  size_t off = 0;
  auto wsf = [&](size_t nfl) {
    float* p = (float*)((char*)d_ws + off);
    off += ((nfl * sizeof(float) + 255) / 256) * 256;
    return p;
  };
  float* featT    = wsf((size_t)HW_ * CF_);             // 16 MB  [hw][c]
  float* loi      = wsf((size_t)HW_ * DLOI);            //  8 MB  [hw][d]
  float* pooled   = wsf((size_t)NE * HOUT_);
  float* featbuf  = wsf((size_t)NE * 2048);             // 64 MB (reused as cat/ecat)
  float* h1buf    = wsf((size_t)NE * HFC_);             // 16 MB (reused as xj)
  float* h2buf    = wsf((size_t)NE * HFC_);             // 16 MB (reused as xi)
  float* h_node   = wsf((size_t)NN * HID_);
  float* h_edge   = wsf((size_t)NE * 256);
  float* hnA      = wsf((size_t)NN * HID_);
  float* hnB      = wsf((size_t)NN * HID_);
  float* heA      = wsf((size_t)NE * 256);
  float* heB      = wsf((size_t)NE * 256);
  float* xfc      = wsf((size_t)NN * HEADS_ * HID_);
  float* alpha    = wsf((size_t)NE * HEADS_ * HID_);
  float* amax     = wsf((size_t)NN * HEADS_ * HID_);
  float* denom    = wsf((size_t)NN * HEADS_ * HID_);
  float* updb     = wsf((size_t)NN * HEADS_ * HID_);
  float* ue       = wsf((size_t)NE * HEADS_ * HID_);
  float* pt_cat   = wsf((size_t)NN * (HID_ + HEADS_ * HID_));
  float* edge_cat = wsf((size_t)NE * (256 + HEADS_ * HID_));
  float* edge_out = wsf((size_t)NE * EOUT_);
  float* mat      = wsf((size_t)NN * NN * 2);
  int*   winner   = (int*)wsf((size_t)NN * NN);
  (void)ws_size; (void)in_sizes; (void)n_in; (void)out_size;

  // ---- Stage 0: transpose features so the loi GEMM is unit-stride in K ---
  transpose_feat_kernel<<<dim3(HW_ / 32, CF_ / 32), 256, 0, s>>>(features, featT);

  // ---- Stage 1: loi[hw, d] = featT[hw,:] @ W + b -------------------------
  gemm(s, featT, CF_, P(P_FCLOI_W), DLOI, P(P_FCLOI_B), loi, DLOI,
       HW_, DLOI, CF_, 0);

  // ---- Stage 2: line pooling -> feat (E x 2048) --------------------------
  pooling_sample_kernel<<<NE, DLOI, 0, s>>>(loi, x, row, col, featbuf);

  // ---- Stage 3: hour MLP --------------------------------------------------
  gemm(s, featbuf, 2048, P(P_H1_W), HFC_, P(P_H1_B), h1buf, HFC_, NE, HFC_, 2048, 1);
  gemm(s, h1buf, HFC_, P(P_H2_W), HFC_, P(P_H2_B), h2buf, HFC_, NE, HFC_, HFC_, 1);
  gemm(s, h2buf, HFC_, P(P_H3_W), HOUT_, P(P_H3_B), pooled, HOUT_, NE, HOUT_, HFC_, 0);

  // ---- Stage 4: node / edge embeddings -----------------------------------
  gemm(s, x, 2, P(P_FCNODE_W), HID_, P(P_FCNODE_B), h_node, HID_, NN, HID_, 2, 0);
  gemm(s, edge_attr, 4, P(P_FCEDGE_W), EHID_, P(P_FCEDGE_B), h_edge, 256,
       NE, EHID_, 4, 0);
  copy2d_kernel<<<EW((long long)NE * HOUT_)>>>(pooled, HOUT_, h_edge, 256,
                                               EHID_, NE, HOUT_);

  // ---- Stage 5: 4 GAT layers ---------------------------------------------
  GatScratch t = { xfc, h2buf, h1buf, featbuf, alpha, amax, denom, updb,
                   ue, pt_cat, edge_cat };
  run_gat(s, d_in, P_CONV_IN,  HID_, HID_, 256,  EHID_, h_node, h_edge, hnA, heA, 0, row, col, t);
  run_gat(s, d_in, P_HIDDEN0,  HID_, HID_, EHID_, EHID_, hnA, heA, hnB, heB, 0, row, col, t);
  run_gat(s, d_in, P_HIDDEN1,  HID_, HID_, EHID_, EHID_, hnB, heB, hnA, heA, 0, row, col, t);
  run_gat(s, d_in, P_CONV_OUT, HID_, ODIM_, EHID_, EOUT_, hnA, heA,
          out, edge_out, 3 /* sigmoid */, row, col, t);

  // ---- Stage 6: symmetric edge matrix ------------------------------------
  fill_int_kernel<<<EW((long long)NN * NN)>>>(winner, -1, (long long)NN * NN);
  winner_kernel<<<EW(NE)>>>(row, col, winner, NE);
  mat_init_kernel<<<EW((long long)NN * NN)>>>(mat);
  mat_set_kernel<<<EW(NE)>>>(edge_out, row, col, winner, mat, NE);
  sym_kernel<<<EW(NE)>>>(mat, row, col, out + (long long)NN * ODIM_, NE);
}
